// Render_75617194213733
// MI455X (gfx1250) — matrix operand order repaired and checked by reading, not compiled, not run on hardware
//
#include <hip/hip_runtime.h>

typedef __attribute__((ext_vector_type(16))) _Float16 v16h;
typedef __attribute__((ext_vector_type(8)))  float    v8f;

#define H_IMG 256
#define W_IMG 256
#define NG    512
#define CHUNK 32
#define NCHUNK (NG / CHUNK)

union V16H { v16h v; _Float16 h[16]; unsigned u[8]; };

// One block = one 16x16 pixel tile.
// Per-pixel sequential alpha blending on the VALU (5 fma + v_exp_f32 + mul +
// fma per Gaussian, opacity folded into the exponent); per-chunk color
// accumulation (weights[16x32] x colors[32x16]) on the matrix pipe via
// v_wmma_f32_16x16x32_f16, accumulating a 16x16 f32 (pixel x channel) tile.
__global__ __launch_bounds__(256) void Render_75617194213733_kernel(
    const float* __restrict__ means, const float* __restrict__ cov,
    const float* __restrict__ opac,  const float* __restrict__ cols,
    float* __restrict__ out)
{
  // [n][0..5] = e0,e1,e2,f0,f1,g (padded to 8 for two ds_load_b128 per gaussian)
  __shared__ float    s_gau[NG][8];                 // 16 KB
  __shared__ unsigned s_bfrag[NCHUNK * 32 * 8];     // 16 KB: per-chunk B fragments
  __shared__ unsigned s_w[2][256 * 16];             // 32 KB: double-buffered weights

  const int t = threadIdx.x;

  // ---- preload: fold inverse covariance, -0.5*log2(e), mean shift and
  // ---- log2(opacity) into 6 polynomial coefficients per gaussian.
  for (int n = t; n < NG; n += 256) {
    float a = cov[4*n+0], b = cov[4*n+1], c = cov[4*n+2], d = cov[4*n+3];
    float inv = 1.0f / (a*d - b*c);
    const float kL = -0.72134752044448170368f;      // -0.5 * log2(e)
    float mx = means[2*n+0], my = means[2*n+1];
    float e0 = kL * d * inv;
    float e1 = kL * (-(b + c)) * inv;
    float e2 = kL * a * inv;
    s_gau[n][0] = e0;
    s_gau[n][1] = e1;
    s_gau[n][2] = e2;
    s_gau[n][3] = -(2.0f*e0*mx + e1*my);            // coeff of x
    s_gau[n][4] = -(2.0f*e2*my + e1*mx);            // coeff of y
    s_gau[n][5] = e0*mx*mx + e1*mx*my + e2*my*my
                + __builtin_amdgcn_logf(opac[n]);   // v_log_f32 is log2
    s_gau[n][6] = 0.0f;
    s_gau[n][7] = 0.0f;
  }

  // ---- preload: prebuilt f16 B fragments (32x16 colors per chunk; RGB in
  // ---- columns 0..2, rest zero), laid out in contiguous K order (k = 16*half + s):
  // ---- lane l holds column n16=l&15, K-half (l>>4)*16, two K per dword.
  for (int d = t * 16, e = d + 16; d < e; ++d) {
    int cc    = d >> 8;           // 256 dwords per chunk
    int l     = (d >> 3) & 31;
    int j     = d & 7;
    int n16   = l & 15;
    int khalf = l >> 4;
    int k     = cc * CHUNK + khalf * 16 + 2 * j;
    union { _Float16 h[2]; unsigned u; } pk;
    pk.h[0] = (n16 < 3) ? (_Float16)cols[k * 3 + n16]       : (_Float16)0.0f;
    pk.h[1] = (n16 < 3) ? (_Float16)cols[(k + 1) * 3 + n16] : (_Float16)0.0f;
    s_bfrag[d] = pk.u;
  }
  __syncthreads();

  const int lx = t & 15, ly = t >> 4;               // thread's own pixel in tile
  const float px = (float)(blockIdx.x * 16 + lx) + 0.5f;
  const float py = (float)(blockIdx.y * 16 + ly) + 0.5f;
  const float xx = px * px, xy = px * py, yy = py * py;
  const float C1 = 1.0f + 1e-7f;

  const int lane  = t & 31;
  const int wave  = t >> 5;
  const int n16   = lane & 15;                      // WMMA row (A) / column (C)
  const int khalf = lane >> 4;                      // which K-half this lane holds

  float T = 1.0f;                                   // transmittance recurrence
  v8f acc[2] = {};                                  // two 16x16 f32 C tiles / wave

  for (int cch = 0; cch < NCHUNK; ++cch) {
    const int base = cch * CHUNK;
    unsigned* wbuf = s_w[cch & 1];

    // ---- sequential scan: 32 alpha-blend steps for this thread's pixel ----
    #pragma unroll
    for (int j = 0; j < 16; ++j) {
      float w01[2];
      #pragma unroll
      for (int s = 0; s < 2; ++s) {
        const float* gp = s_gau[base + 2*j + s];
        float m  = fmaf(gp[0], xx,
                   fmaf(gp[1], xy,
                   fmaf(gp[2], yy,
                   fmaf(gp[3], px,
                   fmaf(gp[4], py, gp[5])))));
        float al = __builtin_amdgcn_exp2f(m);       // alpha (opacity folded in)
        float w  = al * T;
        T        = fmaf(T, C1, -w);                 // T *= (1 - al + 1e-7)
        w01[s]   = w;
      }
      union { _Float16 h[2]; unsigned u; } pk;
      pk.h[0] = (_Float16)w01[0];
      pk.h[1] = (_Float16)w01[1];
      wbuf[t*16 + j] = pk.u;                        // row = pixel, dword = K pair
    }
    __syncthreads();  // waits dscnt==0 per wave; double-buffer makes this the
                      // only barrier needed per chunk.

    // ---- B fragment: two ds_load_b128 from the prebuilt table ----
    V16H bf;
    #pragma unroll
    for (int j = 0; j < 8; ++j) bf.u[j] = s_bfrag[(cch * 32 + lane) * 8 + j];

    // ---- each wave runs WMMA for its two 16-pixel row-groups ----
    // A layout (16-bit 16x32): lanes 0-15 hold K 0-15, lanes 16-31 hold
    // K 16-31 (matching B), two K per VGPR -> dword index in wbuf row is K/2.
    #pragma unroll
    for (int gi = 0; gi < 2; ++gi) {
      int row = (wave*2 + gi)*16 + n16;             // pixel whose weights we carry
      int kd  = khalf * 8;
      V16H af;
      #pragma unroll
      for (int j = 0; j < 4; ++j) af.u[j]     = wbuf[row*16 + kd + j];
      #pragma unroll
      for (int j = 0; j < 4; ++j) af.u[4 + j] = wbuf[row*16 + kd + 4 + j];
      acc[gi] = __builtin_amdgcn_wmma_f32_16x16x32_f16(
          /*neg_a=*/false, af.v, /*neg_b=*/false, bf.v,
          /*c_mod=*/(short)0, acc[gi], /*reuse_a=*/false, /*reuse_b=*/false);
    }
  }

  // ---- writeback: C layout -> VGPR r is M = r + 8*khalf (= lx), N = channel
  if (n16 < 3) {
    #pragma unroll
    for (int gi = 0; gi < 2; ++gi) {
      int pyo = blockIdx.y*16 + wave*2 + gi;        // group index = image row
      #pragma unroll
      for (int r = 0; r < 8; ++r) {
        int ox = blockIdx.x*16 + (r + 8*khalf);
        out[(pyo*W_IMG + ox)*3 + n16] = acc[gi][r];
      }
    }
  }
}

extern "C" void kernel_launch(void* const* d_in, const int* in_sizes, int n_in,
                              void* d_out, int out_size, void* d_ws, size_t ws_size,
                              hipStream_t stream) {
  (void)in_sizes; (void)n_in; (void)out_size; (void)d_ws; (void)ws_size;
  const float* means  = (const float*)d_in[0];   // (512,2)
  const float* cov    = (const float*)d_in[1];   // (512,2,2)
  const float* opac   = (const float*)d_in[2];   // (512,)
  const float* colors = (const float*)d_in[3];   // (512,3)
  float* out = (float*)d_out;                    // (256,256,3) f32

  dim3 grid(W_IMG / 16, H_IMG / 16);
  Render_75617194213733_kernel<<<grid, 256, 0, stream>>>(means, cov, opac, colors, out);
}
